// EnergyGNN_GAT2Conv_22883585753796
// MI455X (gfx1250) — matrix operand order repaired and checked by reading, not compiled, not run on hardware
//
#include <hip/hip_runtime.h>
#include <hip/hip_bf16.h>

typedef __attribute__((ext_vector_type(16))) __bf16 v16bf;
typedef __attribute__((ext_vector_type(8)))  __bf16 v8bf;
typedef __attribute__((ext_vector_type(8)))  float  v8f;

#define HC 128
#define NH 4
#define NC 32
#define NG 64

// float -> bf16 (round-to-nearest-even), stored as u16
static __device__ __forceinline__ unsigned short f2bf(float f) {
  unsigned u = __float_as_uint(f);
  unsigned r = (u + 0x7FFFu + ((u >> 16) & 1u)) >> 16;
  return (unsigned short)r;
}

__global__ void k_f32_to_bf16(const float* __restrict__ in,
                              unsigned short* __restrict__ out, int n) {
  int i = blockIdx.x * blockDim.x + threadIdx.x;
  if (i < n) out[i] = f2bf(in[i]);
}

// WT[n][k] = (bf16) W[k][n]   (W is [128,128] row-major fp32)
__global__ void k_w_transpose_bf16(const float* __restrict__ W,
                                   unsigned short* __restrict__ WT) {
  int i = blockIdx.x * blockDim.x + threadIdx.x;
  if (i < HC * HC) {
    int n = i >> 7, k = i & 127;
    WT[i] = f2bf(W[k * HC + n]);
  }
}

// Dual GEMM: Yl = X@Wl + bl, Yr = X@Wr + br for X[rows,128] bf16.
// 8 waves/block; wave w owns output columns n0 = w*16 and loops 4 M-tiles
// (block covers 64 rows x 128 cols). Both B-fragment sets are hoisted into
// registers once per wave; each A-fragment feeds 2 WMMAs -> 32 WMMAs/wave.
__global__ void k_gemm_dual(const unsigned short* __restrict__ Xbf,
                            const unsigned short* __restrict__ WlT,
                            const unsigned short* __restrict__ WrT,
                            const float* __restrict__ bl,
                            const float* __restrict__ br,
                            float* __restrict__ Yl,
                            float* __restrict__ Yr, int rows) {
  const int wave = threadIdx.x >> 5;
  const int lane = threadIdx.x & 31;
  const int half = lane >> 4;
  const int mn   = lane & 15;
  const int n0   = wave * 16;
  const int m0   = blockIdx.x * 64;

  // B 32x16 fragment, this code's contiguous k order: lanes 0-15 hold K=0..15 of col n, lanes 16-31 K=16..31
  const unsigned short* blrow = WlT + (size_t)(n0 + mn) * HC;
  const unsigned short* brrow = WrT + (size_t)(n0 + mn) * HC;
  v16bf Bl[4], Br[4];
#pragma unroll
  for (int kk = 0; kk < 4; ++kk) {
    Bl[kk] = *(const v16bf*)(blrow + kk * 32 + half * 16);
    Br[kk] = *(const v16bf*)(brrow + kk * 32 + half * 16);
  }
  const float biasl = bl[n0 + mn];
  const float biasr = br[n0 + mn];

#pragma unroll
  for (int mt = 0; mt < 4; ++mt) {
    const int mb = m0 + mt * 16;
    if (mb >= rows) break;                      // wave-uniform
    int ar = mb + mn; if (ar >= rows) ar = rows - 1;  // clamp (tail-safe loads)
    const unsigned short* arow = Xbf + (size_t)ar * HC;

    v8f cl = {0.f, 0.f, 0.f, 0.f, 0.f, 0.f, 0.f, 0.f};
    v8f cr = {0.f, 0.f, 0.f, 0.f, 0.f, 0.f, 0.f, 0.f};
#pragma unroll
    for (int kk = 0; kk < 4; ++kk) {
      const int kb = kk * 32;
      // A 16x32 fragment, contiguous k order: lanes 0-15 K=0..15, lanes 16-31 K=16..31 (matches B)
      union { v16bf v; v8bf h[2]; } a;
      a.v = *(const v16bf*)(arow + kb + half * 16);
      cl = __builtin_amdgcn_wmma_f32_16x16x32_bf16(false, a.v, false, Bl[kk],
                                                   (short)0, cl, false, false);
      cr = __builtin_amdgcn_wmma_f32_16x16x32_bf16(false, a.v, false, Br[kk],
                                                   (short)0, cr, false, false);
    }

    // C VGPR j -> row mb + j + 8*half, col n0 + mn
    float* yl = Yl + (size_t)(mb + 8 * half) * HC + n0 + mn;
    float* yr = Yr + (size_t)(mb + 8 * half) * HC + n0 + mn;
    if (mb + 16 <= rows) {                      // full tile: straight-line stores
#pragma unroll
      for (int j = 0; j < 8; ++j) {
        yl[(size_t)j * HC] = cl[j] + biasl;
        yr[(size_t)j * HC] = cr[j] + biasr;
      }
    } else {                                    // partial tail tile
#pragma unroll
      for (int j = 0; j < 8; ++j) {
        int row = mb + j + 8 * half;
        if (row < rows) {
          Yl[(size_t)row * HC + n0 + mn] = cl[j] + biasl;
          Yr[(size_t)row * HC + n0 + mn] = cr[j] + biasr;
        }
      }
    }
  }
}

__global__ void k_layer_init(float* __restrict__ acc, float* __restrict__ den,
                             unsigned* __restrict__ mx, int N) {
  int i = blockIdx.x * blockDim.x + threadIdx.x;
  if (i < N * HC) acc[i] = 0.f;
  if (i < N * NH) { den[i] = 0.f; mx[i] = 0x007FFFFFu; }  // key(-inf)
}

// Per edge (one wave): m = leaky(xl[src]+xr[dst]+ea@We, 0.2);
// logits[e][h] = dot(m[h], att[h]); atomic segment-max into mx[dst][h].
__global__ void k_edge_logits(const float* __restrict__ xl, const float* __restrict__ xr,
                              const float* __restrict__ ea, const float* __restrict__ We,
                              const float* __restrict__ att,
                              const int* __restrict__ src, const int* __restrict__ dst,
                              float* __restrict__ logits, unsigned* __restrict__ mx, int E) {
  __shared__ float sWe[16 * HC];
  __shared__ float sAtt[HC];
  for (int i = threadIdx.x; i < 16 * HC; i += blockDim.x) sWe[i] = We[i];
  for (int i = threadIdx.x; i < HC; i += blockDim.x) sAtt[i] = att[i];
  __syncthreads();
  const int lane = threadIdx.x & 31;
  const int e = blockIdx.x * (blockDim.x >> 5) + (threadIdx.x >> 5);
  if (e >= E) return;
  const int s = src[e], d = dst[e];
  float eav[16];
#pragma unroll
  for (int q = 0; q < 16; ++q) eav[q] = ea[(size_t)e * 16 + q];
#pragma unroll
  for (int j = 0; j < NH; ++j) {
    const int c = j * NC + lane;           // head j, channel lane
    float em = 0.f;
#pragma unroll
    for (int q = 0; q < 16; ++q) em = fmaf(eav[q], sWe[q * HC + c], em);
    float v = xl[(size_t)s * HC + c] + xr[(size_t)d * HC + c] + em;
    v = v > 0.f ? v : 0.2f * v;
    float t = v * sAtt[c];
#pragma unroll
    for (int off = 16; off > 0; off >>= 1) t += __shfl_down(t, off, 32);
    if (lane == 0) {
      logits[(size_t)e * NH + j] = t;
      unsigned ub = __float_as_uint(t);
      unsigned key = (ub & 0x80000000u) ? ~ub : (ub | 0x80000000u);
      atomicMax(&mx[(size_t)d * NH + j], key);
    }
  }
}

// Decode uint max-keys to floats, sanitize non-finite -> 0 (in place).
__global__ void k_finalize_max(unsigned* __restrict__ mx, int n) {
  int i = blockIdx.x * blockDim.x + threadIdx.x;
  if (i >= n) return;
  unsigned u = mx[i];
  float f = (u & 0x80000000u) ? __uint_as_float(u & 0x7FFFFFFFu)
                              : __uint_as_float(~u);
  if (!(fabsf(f) < __builtin_inff())) f = 0.f;
  ((float*)mx)[i] = f;
}

// a = exp(logit - mx[dst]); den[dst] += a  (logits overwritten with a)
__global__ void k_edge_exp_den(float* __restrict__ logits, const float* __restrict__ mxf,
                               float* __restrict__ den, const int* __restrict__ dst, int E) {
  int i = blockIdx.x * blockDim.x + threadIdx.x;
  if (i >= E * NH) return;
  int e = i >> 2, j = i & 3;
  int d = dst[e];
  float a = __expf(logits[i] - mxf[d * NH + j]);
  logits[i] = a;
  atomicAdd(&den[d * NH + j], a);
}

// acc[dst] += alpha * xl[src]   (one wave per edge, 4 channels per lane)
__global__ void k_edge_aggregate(const float* __restrict__ xl, const float* __restrict__ a,
                                 const float* __restrict__ den,
                                 const int* __restrict__ src, const int* __restrict__ dst,
                                 float* __restrict__ acc, int E) {
  const int lane = threadIdx.x & 31;
  const int e = blockIdx.x * (blockDim.x >> 5) + (threadIdx.x >> 5);
  if (e >= E) return;
  const int s = src[e], d = dst[e];
#pragma unroll
  for (int j = 0; j < NH; ++j) {
    float alpha = a[(size_t)e * NH + j] / (den[d * NH + j] + 1e-16f);
    int c = j * NC + lane;
    atomicAdd(&acc[(size_t)d * HC + c], alpha * xl[(size_t)s * HC + c]);
  }
}

// h = leaky(bn(acc + bias), 0.01); also emit bf16 copy for next layer's GEMM
__global__ void k_post(const float* __restrict__ acc, const float* __restrict__ bias,
                       const float* __restrict__ gamma, const float* __restrict__ beta,
                       const float* __restrict__ mean, const float* __restrict__ var,
                       float* __restrict__ h, unsigned short* __restrict__ hbf, int n) {
  int i = blockIdx.x * blockDim.x + threadIdx.x;
  if (i >= n) return;
  int c = i & (HC - 1);
  float v = acc[i] + bias[c];
  v = (v - mean[c]) * rsqrtf(var[c] + 1e-5f) * gamma[c] + beta[c];
  v = v > 0.f ? v : 0.01f * v;
  h[i] = v;
  hbf[i] = f2bf(v);
}

__global__ void k_pool_init(float* __restrict__ pooled, float* __restrict__ cnt) {
  int i = blockIdx.x * blockDim.x + threadIdx.x;
  if (i < NG * HC) pooled[i] = 0.f;
  if (i < NG) cnt[i] = 0.f;
}

__global__ void k_pool(const float* __restrict__ h, const int* __restrict__ batch,
                       float* __restrict__ pooled, float* __restrict__ cnt, int N) {
  int i = blockIdx.x * blockDim.x + threadIdx.x;
  if (i >= N * HC) return;
  int nid = i >> 7, c = i & 127;
  int g = batch[nid];
  atomicAdd(&pooled[g * HC + c], h[i]);
  if (c == 0) atomicAdd(&cnt[g], 1.f);
}

// One wave per graph: out[g] = relu(mean_pool @ W1 + b1) @ W2 + b2
__global__ void k_head(const float* __restrict__ pooled, const float* __restrict__ cnt,
                       const float* __restrict__ W1, const float* __restrict__ b1,
                       const float* __restrict__ W2, const float* __restrict__ b2,
                       float* __restrict__ out) {
  int g = blockIdx.x, k = threadIdx.x;  // 32 threads
  float inv = 1.f / fmaxf(cnt[g], 1.f);
  float t = b1[k];
  for (int i = 0; i < HC; ++i) t = fmaf(pooled[g * HC + i] * inv, W1[i * NC + k], t);
  t = fmaxf(t, 0.f);
  float v = t * W2[k];
#pragma unroll
  for (int off = 16; off > 0; off >>= 1) v += __shfl_down(v, off, 32);
  if (k == 0) out[g] = v + b2[0];
}

extern "C" void kernel_launch(void* const* d_in, const int* in_sizes, int n_in,
                              void* d_out, int out_size, void* d_ws, size_t ws_size,
                              hipStream_t stream) {
  (void)n_in; (void)out_size; (void)ws_size;
  const float* x     = (const float*)d_in[0];
  const int*   eix   = (const int*)d_in[1];
  const float* ea    = (const float*)d_in[2];
  const int*   batch = (const int*)d_in[3];
  const int N = in_sizes[3];
  const int E = in_sizes[1] / 2;
  const int* src = eix;
  const int* dst = eix + E;

  // Workspace carve (256B aligned)
  char* w = (char*)d_ws;
  auto carve = [&](size_t bytes) { void* p = (void*)w; w += (bytes + 255) & ~(size_t)255; return p; };
  float* h            = (float*)carve((size_t)N * HC * 4);
  float* acc          = (float*)carve((size_t)N * HC * 4);
  unsigned short* xbf = (unsigned short*)carve((size_t)N * HC * 2);
  float* xl           = (float*)carve((size_t)N * HC * 4);
  float* xr           = (float*)carve((size_t)N * HC * 4);
  unsigned short* WlT = (unsigned short*)carve((size_t)HC * HC * 2);
  unsigned short* WrT = (unsigned short*)carve((size_t)HC * HC * 2);
  float* logits       = (float*)carve((size_t)E * NH * 4);
  unsigned* mx        = (unsigned*)carve((size_t)N * NH * 4);
  float* den          = (float*)carve((size_t)N * NH * 4);
  float* pooled       = (float*)carve((size_t)NG * HC * 4);
  float* cnt          = (float*)carve((size_t)NG * 4);

  const int T = 256;
  auto blocks = [](long long n, int t) { return (int)((n + t - 1) / t); };

  k_f32_to_bf16<<<blocks((long long)N * HC, T), T, 0, stream>>>(x, xbf, N * HC);

  for (int l = 0; l < 3; ++l) {
    const float* Wl    = (const float*)d_in[4  + 11 * l];
    const float* bl    = (const float*)d_in[5  + 11 * l];
    const float* Wr    = (const float*)d_in[6  + 11 * l];
    const float* br    = (const float*)d_in[7  + 11 * l];
    const float* We    = (const float*)d_in[8  + 11 * l];
    const float* att   = (const float*)d_in[9  + 11 * l];
    const float* bias  = (const float*)d_in[10 + 11 * l];
    const float* gamma = (const float*)d_in[11 + 11 * l];
    const float* beta  = (const float*)d_in[12 + 11 * l];
    const float* mean  = (const float*)d_in[13 + 11 * l];
    const float* var   = (const float*)d_in[14 + 11 * l];

    k_w_transpose_bf16<<<blocks(HC * HC, T), T, 0, stream>>>(Wl, WlT);
    k_w_transpose_bf16<<<blocks(HC * HC, T), T, 0, stream>>>(Wr, WrT);
    k_gemm_dual<<<blocks(N, 64), 256, 0, stream>>>(xbf, WlT, WrT, bl, br, xl, xr, N);
    k_layer_init<<<blocks((long long)N * HC, T), T, 0, stream>>>(acc, den, mx, N);
    k_edge_logits<<<blocks(E, 8), 256, 0, stream>>>(xl, xr, ea, We, att, src, dst,
                                                    logits, mx, E);
    k_finalize_max<<<blocks((long long)N * NH, T), T, 0, stream>>>(mx, N * NH);
    k_edge_exp_den<<<blocks((long long)E * NH, T), T, 0, stream>>>(logits, (const float*)mx,
                                                                   den, dst, E);
    k_edge_aggregate<<<blocks(E, 8), 256, 0, stream>>>(xl, logits, den, src, dst, acc, E);
    k_post<<<blocks((long long)N * HC, T), T, 0, stream>>>(acc, bias, gamma, beta, mean, var,
                                                           h, xbf, N * HC);
  }

  const float* W1 = (const float*)d_in[37];
  const float* b1 = (const float*)d_in[38];
  const float* W2 = (const float*)d_in[39];
  const float* b2 = (const float*)d_in[40];
  k_pool_init<<<blocks(NG * HC, T), T, 0, stream>>>(pooled, cnt);
  k_pool<<<blocks((long long)N * HC, T), T, 0, stream>>>(h, batch, pooled, cnt, N);
  k_head<<<NG, 32, 0, stream>>>(pooled, cnt, W1, b1, W2, b2, (float*)d_out);
}
